// EdgeConvPoolingFeatures_69200513074022
// MI455X (gfx1250) — matrix operand order repaired and checked by reading, not compiled, not run on hardware
//
#include <hip/hip_runtime.h>
#include <hip/hip_bf16.h>
#include <math.h>

#define KNN 10
#define FLIP_NEGINF 0x007FFFFFu

typedef __attribute__((ext_vector_type(16))) _Float16 v16h;
typedef __attribute__((ext_vector_type(8)))  float    v8f;
typedef __attribute__((ext_vector_type(4)))  unsigned u32x4;

union V16H { v16h v; _Float16 h[16]; u32x4 q[2]; };
union V8F  { v8f  v; float    f[8];  };

// ---- ordered-uint transform for float atomic max ----
__device__ __forceinline__ unsigned fflip(float f) {
  unsigned u = __float_as_uint(f);
  return u ^ (unsigned)(((int)u >> 31) | (int)0x80000000);
}
__device__ __forceinline__ float funflip(unsigned u) {
  unsigned m = ((int)u < 0) ? 0x80000000u : 0xFFFFFFFFu;
  return __uint_as_float(u ^ m);
}

// ====================================================================
// kNN: one thread per query point, insertion top-10 smallest L2 dist
// ====================================================================
__global__ void knn_kernel(const float* __restrict__ x, int B, int N, int D,
                           int* __restrict__ idx) {
  int t = blockIdx.x * blockDim.x + threadIdx.x;
  if (t >= B * N) return;
  int b = t / N, i = t % N;
  const float* xb = x + (size_t)b * N * D;
  float xi[128];
  for (int d = 0; d < D; ++d) xi[d] = xb[(size_t)i * D + d];
  float bd[KNN]; int bi[KNN];
  #pragma unroll
  for (int q = 0; q < KNN; ++q) { bd[q] = INFINITY; bi[q] = 0; }
  for (int j = 0; j < N; ++j) {
    const float* xj = xb + (size_t)j * D;
    __builtin_prefetch(xj + D, 0, 1);
    float dist = 0.f;
    for (int d = 0; d < D; ++d) { float df = xi[d] - xj[d]; dist += df * df; }
    if (dist < bd[KNN - 1]) {
      int p = KNN - 1;
      while (p > 0 && bd[p - 1] > dist) { bd[p] = bd[p - 1]; bi[p] = bi[p - 1]; --p; }
      bd[p] = dist; bi[p] = j;
    }
  }
  #pragma unroll
  for (int q = 0; q < KNN; ++q) idx[(size_t)t * KNN + q] = bi[q];
}

// ====================================================================
// Gather edge features [x_i, x_j - x_i] into zero-padded f16 A matrix
// ====================================================================
__global__ void gather_edges_kernel(const float* __restrict__ x, const int* __restrict__ idx,
                                    int B, int N, int D, int Kp, _Float16* __restrict__ A) {
  int e = blockIdx.x * blockDim.x + threadIdx.x;
  int E = B * N * KNN;
  if (e >= E) return;
  int q = e / KNN;
  int b = q / N, i = q % N;
  int j = idx[e];
  const float* xi = x + ((size_t)b * N + i) * D;
  const float* xj = x + ((size_t)b * N + j) * D;
  _Float16* row = A + (size_t)e * Kp;
  for (int d = 0; d < D; ++d) {
    float a = xi[d];
    row[d]     = (_Float16)a;
    row[D + d] = (_Float16)(xj[d] - a);
  }
  for (int d = 2 * D; d < Kp; ++d) row[d] = (_Float16)0.f;
}

// ====================================================================
// Pack weights W[K,N] (f32) into WMMA B-fragment order, zero-padded.
// Fragments for the NT col-tiles of one k-tile are ADJACENT so the GEMM
// needs a single base pointer + constant immediate offsets (j*1KB):
//   Bp[(((jg*Ktiles + kt)*NT + j)*32 + lane)*16 + i] = f16(W[k,n])
//   k = kt*32 + (i>>3)*16 + (lane>>4)*8 + (i&7), n = (jg*NT + j)*16 + (lane&15)
// (B layout: half h, elems 0..7 -> K=h*8+i, elems 8..15 -> K=16+h*8+i)
// ====================================================================
__global__ void pack_w_kernel(const float* __restrict__ W, int K, int N, int Ktiles, int NT,
                              _Float16* __restrict__ Bp) {
  int t = blockIdx.x * blockDim.x + threadIdx.x;
  int total = (N / 16) * Ktiles * 512;
  if (t >= total) return;
  int i    = t & 15;
  int lane = (t >> 4) & 31;
  int r    = t >> 9;          // fragment index = (jg*Ktiles + kt)*NT + j
  int j    = r % NT;
  r /= NT;
  int kt   = r % Ktiles;
  int jg   = r / Ktiles;
  int k = kt * 32 + ((i >> 3) << 4) + (lane >> 4) * 8 + (i & 7);
  int n = (jg * NT + j) * 16 + (lane & 15);
  Bp[t] = (k < K) ? (_Float16)W[(size_t)k * N + n] : (_Float16)0.f;
}

// ====================================================================
// WMMA GEMM: Y[M,N] = relu?(A[M,lda(f16)] @ Wpacked + bias)
// One wave -> 16x(16*NT) output; single B base pointer with constant
// offsets (no pointer arrays -> pure global_load_b128), all fragments
// loaded before the NT back-to-back v_wmma_f32_16x16x32_f16 ops.
// ====================================================================
template <int NT>
__global__ __launch_bounds__(256)
void gemm_wmma_kernel(const _Float16* __restrict__ A, int lda,
                      const _Float16* __restrict__ Bp,
                      const float* __restrict__ bias,
                      float* __restrict__ Y, int M, int N, int relu) {
  const int wave  = threadIdx.x >> 5;
  const int lane  = threadIdx.x & 31;
  const int tileM = (blockIdx.x * (int)(blockDim.x >> 5) + wave) * 16;
  if (tileM >= M) return;                 // wave-uniform exit; EXEC all-1 below
  const int j0     = blockIdx.y * NT;     // NT 16-col tiles per wave
  const int Ktiles = lda >> 5;
  const int mrow   = lane & 15;
  const int half   = lane >> 4;

  V8F c[NT];
  #pragma unroll
  for (int j = 0; j < NT; ++j)
    #pragma unroll
    for (int i = 0; i < 8; ++i) c[j].f[i] = 0.0f;

  // A layout (16-bit A 16x32): lane m, half h; elems 0..7 -> K=k0+h*8+i,
  //                                            elems 8..15 -> K=k0+16+h*8+i
  const _Float16* arow = A + (size_t)(tileM + mrow) * lda + half * 8;
  const _Float16* bb   = Bp + ((size_t)blockIdx.y * Ktiles * NT * 32 + lane) * 16;

  for (int t = 0; t < Ktiles; ++t) {
    V16H a, b[NT];
    a.q[0] = *(const u32x4*)(arow);
    a.q[1] = *(const u32x4*)(arow + 16);
    #pragma unroll
    for (int j = 0; j < NT; ++j) {
      b[j].q[0] = *(const u32x4*)(bb + j * 512);
      b[j].q[1] = *(const u32x4*)(bb + j * 512 + 8);
    }
    #pragma unroll
    for (int j = 0; j < NT; ++j)
      c[j].v = __builtin_amdgcn_wmma_f32_16x16x32_f16(false, a.v, false, b[j].v,
                                                      (short)0, c[j].v, false, false);
    arow += 32;
    bb   += NT * 512;
  }

  #pragma unroll
  for (int j = 0; j < NT; ++j) {
    const int ncol = (j0 + j) * 16 + (lane & 15);
    const float bv = bias ? bias[ncol] : 0.0f;
    #pragma unroll
    for (int v = 0; v < 8; ++v) {
      int row = tileM + half * 8 + v;     // C/D layout: lanes16-31 -> M=8+v
      float val = c[j].f[v] + bv;
      if (relu) val = fmaxf(val, 0.0f);
      Y[(size_t)row * N + ncol] = val;
    }
  }
}

// ====================================================================
// BatchNorm (training-mode batch stats over all rows)
// ====================================================================
__global__ void zero_f32_kernel(float* p, int n) {
  int i = blockIdx.x * blockDim.x + threadIdx.x;
  if (i < n) p[i] = 0.0f;
}
__global__ void fill_u32_kernel(unsigned* p, int n, unsigned v) {
  int i = blockIdx.x * blockDim.x + threadIdx.x;
  if (i < n) p[i] = v;
}
__global__ void bn_stats_kernel(const float* __restrict__ Y, int M, int C,
                                float* __restrict__ sums) {
  int c = blockIdx.x;
  float s = 0.f, s2 = 0.f;
  for (int r = blockIdx.y * blockDim.x + threadIdx.x; r < M; r += gridDim.y * blockDim.x) {
    float v = Y[(size_t)r * C + c];
    s += v; s2 += v * v;
  }
  atomicAdd(&sums[c], s);
  atomicAdd(&sums[C + c], s2);
}
__global__ void bn_norm_kernel(const float* __restrict__ Y, const float* __restrict__ sums,
                               const float* __restrict__ gamma, const float* __restrict__ beta,
                               int M, int C, float* outF, _Float16* outH) {
  int i = blockIdx.x * blockDim.x + threadIdx.x;
  if (i >= M * C) return;
  int c = i % C;
  float invM = 1.0f / (float)M;
  float mean = sums[c] * invM;
  float var  = sums[C + c] * invM - mean * mean;
  float v = (Y[i] - mean) * rsqrtf(var + 1e-5f) * gamma[c] + beta[c];
  if (outH) outH[i] = (_Float16)v; else outF[i] = v;
}

// max over k neighbors: Y[Mn*KNN, C] -> X[Mn, C]
__global__ void edge_max_kernel(const float* __restrict__ Y, int Mn, int C,
                                float* __restrict__ X) {
  int i = blockIdx.x * blockDim.x + threadIdx.x;
  if (i >= Mn * C) return;
  int node = i / C, c = i % C;
  const float* y = Y + ((size_t)node * KNN) * C + c;
  float m = -INFINITY;
  #pragma unroll
  for (int q = 0; q < KNN; ++q) m = fmaxf(m, y[(size_t)q * C]);
  X[i] = m;
}

// ====================================================================
// ASAP pooling pieces
// ====================================================================
__global__ void segmax_feat_kernel(const float* __restrict__ X, const int* __restrict__ idx,
                                   int B, int N, int F, unsigned* __restrict__ xqu) {
  int t = blockIdx.x * blockDim.x + threadIdx.x;
  int E = B * N * KNN;
  if (t >= E * F) return;
  int e = t / F, f = t % F;
  int q = e / KNN;
  int b = q / N;
  int dst = b * N + idx[e];
  atomicMax(&xqu[(size_t)dst * F + f], fflip(X[(size_t)q * F + f]));
}
__global__ void unflip_half_kernel(const unsigned* __restrict__ u, int n,
                                   _Float16* __restrict__ h) {
  int i = blockIdx.x * blockDim.x + threadIdx.x;
  if (i < n) h[i] = (_Float16)funflip(u[i]);
}
__global__ void att_score_kernel(const float* __restrict__ xq2, const float* __restrict__ X,
                                 const int* __restrict__ idx, const float* __restrict__ Wa,
                                 const float* __restrict__ ba, int B, int N, int F,
                                 float* __restrict__ s) {
  int e = blockIdx.x * blockDim.x + threadIdx.x;
  int E = B * N * KNN;
  if (e >= E) return;
  int q = e / KNN;
  int b = q / N;
  int dst = b * N + idx[e];
  const float* xd = xq2 + (size_t)dst * F;
  const float* xs = X   + (size_t)q   * F;
  float acc = ba[0];
  for (int f = 0; f < F; ++f) acc += xd[f] * Wa[f];
  for (int f = 0; f < F; ++f) acc += xs[f] * Wa[F + f];
  s[e] = (acc >= 0.f) ? acc : 0.2f * acc;   // leaky_relu(0.2)
}
__global__ void seg_smax_kernel(const float* __restrict__ s, const int* __restrict__ idx,
                                int B, int N, unsigned* __restrict__ smaxu) {
  int e = blockIdx.x * blockDim.x + threadIdx.x;
  int E = B * N * KNN;
  if (e >= E) return;
  int q = e / KNN; int b = q / N;
  atomicMax(&smaxu[b * N + idx[e]], fflip(s[e]));
}
__global__ void seg_expsum_kernel(const float* __restrict__ s, const int* __restrict__ idx,
                                  const unsigned* __restrict__ smaxu, int B, int N,
                                  float* __restrict__ eexp, float* __restrict__ ssum) {
  int e = blockIdx.x * blockDim.x + threadIdx.x;
  int E = B * N * KNN;
  if (e >= E) return;
  int q = e / KNN; int b = q / N;
  int dst = b * N + idx[e];
  float ee = __expf(s[e] - funflip(smaxu[dst]));
  eexp[e] = ee;
  atomicAdd(&ssum[dst], ee);
}
__global__ void seg_weighted_kernel(const float* __restrict__ X, const int* __restrict__ idx,
                                    const float* __restrict__ eexp, const float* __restrict__ ssum,
                                    int B, int N, int F, float* __restrict__ xc) {
  int e = blockIdx.x * blockDim.x + threadIdx.x;
  int E = B * N * KNN;
  if (e >= E) return;
  int q = e / KNN; int b = q / N;
  int dst = b * N + idx[e];
  float alpha = eexp[e] / ssum[dst];
  const float* xs = X + (size_t)q * F;
  float* xd = xc + (size_t)dst * F;
  for (int f = 0; f < F; ++f) atomicAdd(&xd[f], xs[f] * alpha);
}
__global__ void leconv_node_kernel(const float* __restrict__ xc, const float* __restrict__ W1,
                                   const float* __restrict__ b1, const float* __restrict__ W2,
                                   int M, int F, float* __restrict__ a, float* __restrict__ bb) {
  int m = blockIdx.x * blockDim.x + threadIdx.x;
  if (m >= M) return;
  const float* xr = xc + (size_t)m * F;
  float s1 = b1[0], s2 = 0.f;
  for (int f = 0; f < F; ++f) { s1 += xr[f] * W1[f]; s2 += xr[f] * W2[f]; }
  a[m] = s1; bb[m] = s2;
}
__global__ void leconv_edge_kernel(const float* __restrict__ a, const float* __restrict__ bb,
                                   const int* __restrict__ idx, int B, int N,
                                   float* __restrict__ agg) {
  int e = blockIdx.x * blockDim.x + threadIdx.x;
  int E = B * N * KNN;
  if (e >= E) return;
  int q = e / KNN; int b = q / N;
  int dst = b * N + idx[e];
  atomicAdd(&agg[dst], a[q] - bb[dst]);
}
__global__ void leconv_fit_kernel(const float* __restrict__ xc, const float* __restrict__ agg,
                                  const float* __restrict__ W3, const float* __restrict__ b3,
                                  int M, int F, float* __restrict__ fit) {
  int m = blockIdx.x * blockDim.x + threadIdx.x;
  if (m >= M) return;
  const float* xr = xc + (size_t)m * F;
  float s = b3[0];
  for (int f = 0; f < F; ++f) s += xr[f] * W3[f];
  s += agg[m];
  fit[m] = 1.0f / (1.0f + __expf(-s));
}
// exact top-k via stable rank (matches lax.top_k ordering + tie-break)
__global__ void topk_pool_kernel(const float* __restrict__ fit, const float* __restrict__ xc,
                                 int N, int F, int kk, float* __restrict__ Xout) {
  int b = blockIdx.x;
  extern __shared__ float sf[];
  for (int i = threadIdx.x; i < N; i += blockDim.x) sf[i] = fit[(size_t)b * N + i];
  __syncthreads();
  for (int i = threadIdx.x; i < N; i += blockDim.x) {
    float v = sf[i];
    int rank = 0;
    for (int j = 0; j < N; ++j) {
      float u = sf[j];
      rank += (u > v) || (u == v && j < i);
    }
    if (rank < kk) {
      const float* src = xc + ((size_t)b * N + i) * F;
      float* dst = Xout + ((size_t)b * kk + rank) * F;
      for (int f = 0; f < F; ++f) dst[f] = src[f] * v;
    }
  }
}

// global max pool over points
__global__ void global_max_kernel(const float* __restrict__ X, int N, int C,
                                  float* __restrict__ out) {
  int b = blockIdx.x;
  int c = blockIdx.y * blockDim.x + threadIdx.x;
  if (c >= C) return;
  float m = -INFINITY;
  for (int n = 0; n < N; ++n) m = fmaxf(m, X[((size_t)b * N + n) * C + c]);
  out[(size_t)b * C + c] = m;
}
__global__ void to_half_kernel(const float* __restrict__ x, int n, _Float16* __restrict__ h) {
  int i = blockIdx.x * blockDim.x + threadIdx.x;
  if (i < n) h[i] = (_Float16)x[i];
}

// ====================================================================
// Host orchestration
// ====================================================================
struct MLPLayer { const float *W, *b, *gamma, *beta; int cin, cout; };
struct PoolP { const float *Wl, *bl, *Wa, *ba, *W1, *b1, *W2, *W3, *b3; };
struct WS {
  _Float16* A16; float* Yf; int* idx; float* sums; _Float16* Wpk;
  unsigned* xqu; _Float16* xqh; float* xq2; float* xc;
  float* sArr; float* eexp; float* ssum; unsigned* smaxu;
  float* aArr; float* bbArr; float* agg; float* fit;
  float *X1, *P1, *X2, *P2, *X3, *xm; _Float16* xmh;
};

static inline int cdiv(int a, int b) { return (a + b - 1) / b; }

static void launch_gemm(hipStream_t s, const WS& w, const _Float16* A, int lda,
                        const float* Wf, const float* bias, float* Y,
                        int M, int K, int N, int relu) {
  int Ktiles = lda / 32;
  int total = (N / 16) * Ktiles * 512;
  int NT = (N % 64 == 0) ? 4 : 2;
  pack_w_kernel<<<cdiv(total, 256), 256, 0, s>>>(Wf, K, N, Ktiles, NT, w.Wpk);
  if (NT == 4) {
    dim3 grid(cdiv(M / 16, 8), N / 64);
    gemm_wmma_kernel<4><<<grid, dim3(256), 0, s>>>(A, lda, w.Wpk, bias, Y, M, N, relu);
  } else {
    dim3 grid(cdiv(M / 16, 8), N / 32);
    gemm_wmma_kernel<2><<<grid, dim3(256), 0, s>>>(A, lda, w.Wpk, bias, Y, M, N, relu);
  }
}

static void run_edge_conv(hipStream_t s, const WS& w, const float* Xin,
                          int B, int N, int D, const MLPLayer* L, float* Xout) {
  int M = B * N, E = M * KNN;
  knn_kernel<<<cdiv(M, 256), 256, 0, s>>>(Xin, B, N, D, w.idx);
  int Kp = ((2 * D) + 31) / 32 * 32;
  gather_edges_kernel<<<cdiv(E, 256), 256, 0, s>>>(Xin, w.idx, B, N, D, Kp, w.A16);
  int curK = 2 * D, curLda = Kp;
  for (int l = 0; l < 3; ++l) {
    int C = L[l].cout;
    launch_gemm(s, w, w.A16, curLda, L[l].W, L[l].b, w.Yf, E, curK, C, 1);
    zero_f32_kernel<<<cdiv(2 * C, 256), 256, 0, s>>>(w.sums, 2 * C);
    bn_stats_kernel<<<dim3(C, 64), 256, 0, s>>>(w.Yf, E, C, w.sums);
    bool last = (l == 2);
    bn_norm_kernel<<<cdiv(E * C, 256), 256, 0, s>>>(
        w.Yf, w.sums, L[l].gamma, L[l].beta, E, C,
        last ? w.Yf : nullptr, last ? nullptr : w.A16);
    curK = C; curLda = C;
  }
  edge_max_kernel<<<cdiv(M * L[2].cout, 256), 256, 0, s>>>(w.Yf, M, L[2].cout, Xout);
}

static void run_asap(hipStream_t s, const WS& w, const float* X,
                     int B, int N, int F, const PoolP& p, float* Xout) {
  int M = B * N, E = M * KNN, kk = N / 2;
  knn_kernel<<<cdiv(M, 256), 256, 0, s>>>(X, B, N, F, w.idx);
  fill_u32_kernel<<<cdiv(M * F, 256), 256, 0, s>>>(w.xqu, M * F, FLIP_NEGINF);
  segmax_feat_kernel<<<cdiv(E * F, 256), 256, 0, s>>>(X, w.idx, B, N, F, w.xqu);
  unflip_half_kernel<<<cdiv(M * F, 256), 256, 0, s>>>(w.xqu, M * F, w.xqh);
  launch_gemm(s, w, w.xqh, F, p.Wl, p.bl, w.xq2, M, F, F, 0);
  att_score_kernel<<<cdiv(E, 256), 256, 0, s>>>(w.xq2, X, w.idx, p.Wa, p.ba, B, N, F, w.sArr);
  fill_u32_kernel<<<cdiv(M, 256), 256, 0, s>>>(w.smaxu, M, FLIP_NEGINF);
  seg_smax_kernel<<<cdiv(E, 256), 256, 0, s>>>(w.sArr, w.idx, B, N, w.smaxu);
  zero_f32_kernel<<<cdiv(M, 256), 256, 0, s>>>(w.ssum, M);
  seg_expsum_kernel<<<cdiv(E, 256), 256, 0, s>>>(w.sArr, w.idx, w.smaxu, B, N, w.eexp, w.ssum);
  zero_f32_kernel<<<cdiv(M * F, 256), 256, 0, s>>>(w.xc, M * F);
  seg_weighted_kernel<<<cdiv(E, 256), 256, 0, s>>>(X, w.idx, w.eexp, w.ssum, B, N, F, w.xc);
  leconv_node_kernel<<<cdiv(M, 256), 256, 0, s>>>(w.xc, p.W1, p.b1, p.W2, M, F, w.aArr, w.bbArr);
  zero_f32_kernel<<<cdiv(M, 256), 256, 0, s>>>(w.agg, M);
  leconv_edge_kernel<<<cdiv(E, 256), 256, 0, s>>>(w.aArr, w.bbArr, w.idx, B, N, w.agg);
  leconv_fit_kernel<<<cdiv(M, 256), 256, 0, s>>>(w.xc, w.agg, p.W3, p.b3, M, F, w.fit);
  topk_pool_kernel<<<B, 256, N * sizeof(float), s>>>(w.fit, w.xc, N, F, kk, Xout);
}

extern "C" void kernel_launch(void* const* d_in, const int* in_sizes, int n_in,
                              void* d_out, int out_size, void* d_ws, size_t ws_size,
                              hipStream_t stream) {
  (void)in_sizes; (void)n_in; (void)out_size; (void)ws_size;
  const float* positions = (const float*)d_in[0];
  auto F32 = [&](int i) { return (const float*)d_in[i]; };
  MLPLayer c1[3] = {
    {F32(1),  F32(2),  F32(3),  F32(4),  6,  64},
    {F32(5),  F32(6),  F32(7),  F32(8),  64, 64},
    {F32(9),  F32(10), F32(11), F32(12), 64, 32}};
  PoolP p1 = {F32(13), F32(14), F32(15), F32(16), F32(17), F32(18), F32(19), F32(20), F32(21)};
  MLPLayer c2[3] = {
    {F32(22), F32(23), F32(24), F32(25), 64,  128},
    {F32(26), F32(27), F32(28), F32(29), 128, 128},
    {F32(30), F32(31), F32(32), F32(33), 128, 128}};
  PoolP p2 = {F32(34), F32(35), F32(36), F32(37), F32(38), F32(39), F32(40), F32(41), F32(42)};
  MLPLayer c3[3] = {
    {F32(43), F32(44), F32(45), F32(46), 256, 256},
    {F32(47), F32(48), F32(49), F32(50), 256, 256},
    {F32(51), F32(52), F32(53), F32(54), 256, 256}};
  const float* Wout = F32(55);
  const float* bout = F32(56);

  const int B = 16, N1 = 2048, N2 = 1024, N3 = 512;
  const int E1 = B * N1 * KNN;                // 327680
  const int Mmax = B * N1;                    // 32768
  const size_t MFmax = (size_t)B * N2 * 128;  // 2,097,152 (stage-2 ASAP)

  // carve workspace
  char* base = (char*)d_ws;
  size_t off = 0;
  auto carve = [&](size_t bytes) -> void* {
    void* p = base + off;
    off = (off + bytes + 255) & ~(size_t)255;
    return p;
  };
  WS w;
  w.A16   = (_Float16*)carve((size_t)B * N3 * KNN * 512 * sizeof(_Float16)); // 84MB (conv3 A)
  w.Yf    = (float*)   carve((size_t)E1 * 64 * sizeof(float));               // 84MB
  w.idx   = (int*)     carve((size_t)E1 * sizeof(int));
  w.sums  = (float*)   carve(1024 * sizeof(float));
  w.Wpk   = (_Float16*)carve((size_t)256 * 512 * sizeof(_Float16));          // max Kp*N (Wout)
  w.xqu   = (unsigned*)carve(MFmax * sizeof(unsigned));
  w.xqh   = (_Float16*)carve(MFmax * sizeof(_Float16));
  w.xq2   = (float*)   carve(MFmax * sizeof(float));
  w.xc    = (float*)   carve(MFmax * sizeof(float));
  w.sArr  = (float*)   carve((size_t)E1 * sizeof(float));
  w.eexp  = (float*)   carve((size_t)E1 * sizeof(float));
  w.ssum  = (float*)   carve((size_t)Mmax * sizeof(float));
  w.smaxu = (unsigned*)carve((size_t)Mmax * sizeof(unsigned));
  w.aArr  = (float*)   carve((size_t)Mmax * sizeof(float));
  w.bbArr = (float*)   carve((size_t)Mmax * sizeof(float));
  w.agg   = (float*)   carve((size_t)Mmax * sizeof(float));
  w.fit   = (float*)   carve((size_t)Mmax * sizeof(float));
  w.X1    = (float*)   carve((size_t)B * N1 * 32  * sizeof(float));
  w.P1    = (float*)   carve((size_t)B * N2 * 32  * sizeof(float));
  w.X2    = (float*)   carve((size_t)B * N2 * 128 * sizeof(float));
  w.P2    = (float*)   carve((size_t)B * N3 * 128 * sizeof(float));
  w.X3    = (float*)   carve((size_t)B * N3 * 256 * sizeof(float));
  w.xm    = (float*)   carve((size_t)B * 256 * sizeof(float));
  w.xmh   = (_Float16*)carve((size_t)B * 256 * sizeof(_Float16));

  // pipeline
  run_edge_conv(stream, w, positions, B, N1, 3,   c1, w.X1);  // [B,2048,32]
  run_asap     (stream, w, w.X1,      B, N1, 32,  p1, w.P1);  // [B,1024,32]
  run_edge_conv(stream, w, w.P1,      B, N2, 32,  c2, w.X2);  // [B,1024,128]
  run_asap     (stream, w, w.X2,      B, N2, 128, p2, w.P2);  // [B,512,128]
  run_edge_conv(stream, w, w.P2,      B, N3, 128, c3, w.X3);  // [B,512,256]

  global_max_kernel<<<dim3(B, 1), 256, 0, stream>>>(w.X3, N3, 256, w.xm);
  to_half_kernel<<<cdiv(B * 256, 256), 256, 0, stream>>>(w.xm, B * 256, w.xmh);
  launch_gemm(stream, w, w.xmh, 256, Wout, bout, (float*)d_out, B, 256, 512, 0);
}
